// Recurrence_9122510537121
// MI455X (gfx1250) — compile-verified, matrix operand order checked by reading, not run on hardware
//
#include <hip/hip_runtime.h>

// ---------------------------------------------------------------------------
// Problem constants (from reference)
// ---------------------------------------------------------------------------
#define T_    64
#define N_    256
#define OBS_  256
#define HID_  1024
#define EMB_  512
#define NACT_ 16
#define HX_   2578      // 1 + 16 + 1 + 512 + 2048
#define G3_   3072      // 3*HID
#define KMI_  544       // mi K (528) padded to multiple of 32

typedef __bf16 bf16_t;
typedef __attribute__((ext_vector_type(16))) __bf16 v16bf;
typedef __attribute__((ext_vector_type(8)))  float  v8f;
typedef __attribute__((ext_vector_type(4)))  float  f32x4;

#ifndef __has_builtin
#define __has_builtin(x) 0
#endif
#if __has_builtin(__builtin_amdgcn_global_load_async_to_lds_b128)
#define ASYNC_COPY 1
#else
#define ASYNC_COPY 0
#endif

// ---------------------------------------------------------------------------
// WMMA helpers (CDNA5: v_wmma_f32_16x16x32_bf16, wave32)
// ---------------------------------------------------------------------------
__device__ __forceinline__ v8f wmma_bf16(v16bf a, v16bf b, v8f c) {
    // 8 args: (neg_a, A, neg_b, B, c_mod, C, reuse_a, reuse_b)
    return __builtin_amdgcn_wmma_f32_16x16x32_bf16(false, a, false, b,
                                                   (short)0, c, false, false);
}

// Load a 16x32 bf16 fragment (A layout; identical per-lane pattern for B of
// x @ W.T since W is stored (out,in) row-major).  ISA layout:
//   lanes 0-15 : row = lane,    K chunks [0,8) and [16,24)
//   lanes16-31 : row = lane-16, K chunks [8,16) and [24,32)
__device__ __forceinline__ v16bf load_frag(const bf16_t* __restrict__ p0, int ld) {
    const int lane = threadIdx.x & 31;
    const int r    = lane & 15;
    const int kc   = (lane >> 4) << 3;            // 0 or 8
    const bf16_t* p = p0 + (long)r * ld + kc;
    union { v16bf v; f32x4 q[2]; } u;
    u.q[0] = *(const f32x4*)(p);                  // K[kc .. kc+8)
    u.q[1] = *(const f32x4*)(p + 16);             // K[kc+16 .. kc+24)
    return u.v;
}

__device__ __forceinline__ v16bf relu_frag(v16bf a) {
#pragma unroll
    for (int i = 0; i < 16; ++i) {
        float f = (float)a[i];
        a[i] = (bf16_t)(f > 0.f ? f : 0.f);
    }
    return a;
}

// ---------------------------------------------------------------------------
// CDNA5 async global->LDS copy (ASYNCcnt-tracked), with safe fallback
// ---------------------------------------------------------------------------
#if ASYNC_COPY
// builtin signature (from hipcc diagnostic): first param is
//   '__attribute__((vector_size(16))) int __device__ *'  (v4i32 in AS1)
// second is the LDS-side v4i32 pointer (AS3), then imm offset, imm cpol.
typedef int i32x4g __attribute__((vector_size(16)));
typedef __attribute__((address_space(1))) i32x4g* as1_v4i_t;
typedef __attribute__((address_space(3))) i32x4g* as3_v4i_t;

__device__ __forceinline__ void async_b128(const bf16_t* g, bf16_t* l) {
    // generic->AS1 and generic->AS3 via integer reinterpret; flat LDS
    // addresses map to the LDS offset in their low 32 bits (ISA 10.2).
    as1_v4i_t gp = (as1_v4i_t)(unsigned long long)g;
    as3_v4i_t lp = (as3_v4i_t)(unsigned int)(unsigned long long)l;
    __builtin_amdgcn_global_load_async_to_lds_b128(gp, lp, 0, 0);
}
#endif

template<int NN>
__device__ __forceinline__ void wait_async() {
#if ASYNC_COPY
#if __has_builtin(__builtin_amdgcn_s_wait_asynccnt)
    __builtin_amdgcn_s_wait_asynccnt(NN);
#else
    asm volatile("s_wait_asynccnt %0" :: "n"(NN) : "memory");
#endif
#endif
}

// ---------------------------------------------------------------------------
// Generic WMMA GEMM (direct global fragment loads):
//   C(MxN) = act(A(MxK) @ B(NxK)^T + bias)
// block = 128 threads = 4 waves; block tile 64x64; wave tile 32x32 (4 accums)
// Requires M%64==0, N%64==0, K%32==0.  Writes f32 and/or bf16 outputs.
// ---------------------------------------------------------------------------
template<int RELU_A, int RELU_OUT>
__global__ __launch_bounds__(128)
void gemm_wmma_kernel(const bf16_t* __restrict__ A, int lda,
                      const bf16_t* __restrict__ B, int ldb,
                      const float*  __restrict__ bias,
                      float*        __restrict__ Cf, int ldcf,
                      bf16_t*       __restrict__ Cb, int ldcb,
                      int K)
{
    const int  lane = threadIdx.x & 31;
    const int  wave = threadIdx.x >> 5;
    const long row0 = (long)blockIdx.y * 64 + (wave >> 1) * 32;
    const long col0 = (long)blockIdx.x * 64 + (wave & 1)  * 32;

    const bf16_t* pa0 = A + row0 * lda;
    const bf16_t* pa1 = pa0 + 16L * lda;
    const bf16_t* pb0 = B + col0 * ldb;
    const bf16_t* pb1 = pb0 + 16L * ldb;

    v8f c00 = {}, c01 = {}, c10 = {}, c11 = {};
    for (int k = 0; k < K; k += 32) {
        v16bf a0 = load_frag(pa0 + k, lda);
        v16bf a1 = load_frag(pa1 + k, lda);
        if (RELU_A) { a0 = relu_frag(a0); a1 = relu_frag(a1); }
        v16bf b0 = load_frag(pb0 + k, ldb);
        v16bf b1 = load_frag(pb1 + k, ldb);
        c00 = wmma_bf16(a0, b0, c00);
        c01 = wmma_bf16(a0, b1, c01);
        c10 = wmma_bf16(a1, b0, c10);
        c11 = wmma_bf16(a1, b1, c11);
        // stream L2-resident weights ahead (emits global_prefetch_b8)
        __builtin_prefetch(pb0 + k + 256, 0, 1);
        __builtin_prefetch(pb1 + k + 256, 0, 1);
    }

    // C/D layout: VGPR v, lanes 0-15 -> M=v, lanes 16-31 -> M=v+8; N = lane&15
    const int   cn  = lane & 15;
    const int   mo  = (lane >> 4) * 8;
    const float bv0 = bias ? bias[col0 + cn]      : 0.f;
    const float bv1 = bias ? bias[col0 + 16 + cn] : 0.f;

    auto emit = [&](const v8f& c, long r0, long cc, float bv) {
#pragma unroll
        for (int v = 0; v < 8; ++v) {
            long  m = r0 + mo + v;
            float x = c[v] + bv;
            if (RELU_OUT) x = fmaxf(x, 0.f);
            if (Cf) Cf[m * (long)ldcf + cc + cn] = x;
            if (Cb) Cb[m * (long)ldcb + cc + cn] = (bf16_t)x;
        }
    };
    emit(c00, row0,      col0,      bv0);
    emit(c01, row0,      col0 + 16, bv1);
    emit(c10, row0 + 16, col0,      bv0);
    emit(c11, row0 + 16, col0 + 16, bv1);
}

// ---------------------------------------------------------------------------
// LDS-staged WMMA GEMM for the sequential GRU-gate GEMMs.
// Double-buffered 64x32 A and B tiles staged cooperatively via
// GLOBAL_LOAD_ASYNC_TO_LDS_B128 (fallback: load + ds_store_b128), consumed
// with ds_load_b128 fragment reads.  f32 output + bias only.
// ---------------------------------------------------------------------------
__global__ __launch_bounds__(128)
void gemm_lds_kernel(const bf16_t* __restrict__ A, int lda,
                     const bf16_t* __restrict__ B, int ldb,
                     const float*  __restrict__ bias,
                     float*        __restrict__ Cf, int ldcf,
                     int K)
{
    __shared__ bf16_t sA[2][64 * 32];
    __shared__ bf16_t sB[2][64 * 32];

    const int  tid  = threadIdx.x;
    const int  lane = tid & 31;
    const int  wave = tid >> 5;
    const int  wr   = wave >> 1;
    const int  wc   = wave & 1;
    const long rowB = (long)blockIdx.y * 64;      // block row base
    const long colB = (long)blockIdx.x * 64;      // block col base

    // cooperative tile copy: 512 x 16B chunks (A: 256, B: 256), 4 per thread
    auto copy_tile = [&](int buf, int kk) {
        const int kbase = kk * 32;
#pragma unroll
        for (int c = tid; c < 512; c += 128) {
            const int isB  = c >> 8;
            const int cc   = c & 255;
            const int row  = cc >> 2;
            const int part = cc & 3;
            const bf16_t* g = (isB ? B + (colB + row) * (long)ldb
                                   : A + (rowB + row) * (long)lda)
                              + kbase + part * 8;
            bf16_t* l = (isB ? sB[buf] : sA[buf]) + row * 32 + part * 8;
#if ASYNC_COPY
            async_b128(g, l);
#else
            *(f32x4*)l = *(const f32x4*)g;        // global load + ds_store
#endif
        }
    };

    v8f c00 = {}, c01 = {}, c10 = {}, c11 = {};
    const int KS = K >> 5;

    copy_tile(0, 0);
    for (int kk = 0; kk < KS; ++kk) {
        if (kk + 1 < KS) {
            copy_tile((kk + 1) & 1, kk + 1);
            wait_async<4>();                      // previous batch complete
        } else {
            wait_async<0>();
        }
        __syncthreads();                          // s_barrier_signal/wait

        const bf16_t* tA = sA[kk & 1];
        const bf16_t* tB = sB[kk & 1];
        v16bf a0 = load_frag(tA + (wr * 32 +  0) * 32, 32);
        v16bf a1 = load_frag(tA + (wr * 32 + 16) * 32, 32);
        v16bf b0 = load_frag(tB + (wc * 32 +  0) * 32, 32);
        v16bf b1 = load_frag(tB + (wc * 32 + 16) * 32, 32);
        c00 = wmma_bf16(a0, b0, c00);
        c01 = wmma_bf16(a0, b1, c01);
        c10 = wmma_bf16(a1, b0, c10);
        c11 = wmma_bf16(a1, b1, c11);

        __syncthreads();                          // tiles consumed -> reusable
    }

    const int   cn   = lane & 15;
    const int   mo   = (lane >> 4) * 8;
    const long  row0 = rowB + wr * 32;
    const long  col0 = colB + wc * 32;
    const float bv0  = bias[col0 + cn];
    const float bv1  = bias[col0 + 16 + cn];

    auto emit = [&](const v8f& c, long r0, long cc, float bv) {
#pragma unroll
        for (int v = 0; v < 8; ++v)
            Cf[(r0 + mo + v) * (long)ldcf + cc + cn] = c[v] + bv;
    };
    emit(c00, row0,      col0,      bv0);
    emit(c01, row0,      col0 + 16, bv1);
    emit(c10, row0 + 16, col0,      bv0);
    emit(c11, row0 + 16, col0 + 16, bv1);
}

// ---------------------------------------------------------------------------
// Actor/value head: logits = X @ Wa^T + ba (16 cols), v = X @ Wc^T + bc.
// Whead is 32x1024 bf16: rows 0-15 = Wa, row 16 = Wc, rest zero.
// ---------------------------------------------------------------------------
__device__ __forceinline__ float u01_hash(unsigned s) {
    s ^= s >> 17; s *= 0xed5ad4bbu; s ^= s >> 11; s *= 0xac4c1b51u;
    s ^= s >> 15; s *= 0x31848babu; s ^= s >> 14;
    return (float)(s >> 8) * (1.0f / 16777216.0f);
}

__global__ __launch_bounds__(128)
void head_kernel(const bf16_t* __restrict__ X,      // (T*N) x 1024
                 const bf16_t* __restrict__ Whead,  // 32 x 1024
                 const float*  __restrict__ ba, const float* __restrict__ bc,
                 const float*  __restrict__ inputs, // (T,N,257)
                 const bf16_t* __restrict__ EactB,  // 16x16
                 float*        __restrict__ out,    // (T+1,N,2578)
                 bf16_t*       __restrict__ miE)    // (T*N) x 16
{
    const int  lane = threadIdx.x & 31;
    const int  wave = threadIdx.x >> 5;
    const long r0   = ((long)blockIdx.x * 4 + wave) * 16;

    v8f cl = {}, cv = {};
    const bf16_t* pa = X + r0 * HID_;
    for (int k = 0; k < HID_; k += 32) {
        v16bf a  = load_frag(pa + k, HID_);
        v16bf b0 = load_frag(Whead + k, HID_);              // cols 0-15: Wa
        v16bf b1 = load_frag(Whead + 16L * HID_ + k, HID_); // col 16: Wc
        cl = wmma_bf16(a, b0, cl);
        cv = wmma_bf16(a, b1, cv);
    }

    const int   n    = lane & 15;
    const int   half = lane >> 4;
    const float ba_n = ba[n];
    const float bc0  = bc[0];

#pragma unroll
    for (int v = 0; v < 8; ++v) {
        long row  = r0 + half * 8 + v;                // global (t*N + nb)
        int  t    = (int)(row / N_);
        int  nb   = (int)(row % N_);
        long base = row * (long)HX_;

        float logit = cl[v] + ba_n;
        float mx = logit;
        for (int d = 8; d >= 1; d >>= 1) mx = fmaxf(mx, __shfl_xor(mx, d, 16));
        float e = __expf(logit - mx);
        float s = e;
        for (int d = 8; d >= 1; d >>= 1) s += __shfl_xor(s, d, 16);
        float p = e / s;
        out[base + 1 + n] = p;                        // Ps
        if (n == 0) out[base + 17] = cv[v] + bc0;     // Vs (col 16 overall)

        float cdf = p;
        for (int d = 1; d < 16; d <<= 1) {
            float tv = __shfl_up(cdf, (unsigned)d, 16);
            if (n >= d) cdf += tv;
        }
        float u = u01_hash(0x9e3779b9u ^ (unsigned)(t * 0x85ebca6b) ^ (unsigned)(nb * 0xc2b2ae35));
        unsigned long long bal = __ballot(u >= cdf);
        unsigned hm = (unsigned)((bal >> (half * 16)) & 0xFFFFull);
        int samp = __popc(hm); if (samp > 15) samp = 15;

        int a_in = (int)inputs[((long)t * N_ + nb) * (OBS_ + 1) + OBS_];
        int act  = (a_in < 0) ? samp : a_in;
        out[base + 0] = (float)act;                   // As
        miE[((long)t * N_ + nb) * NACT_ + n] = EactB[act * NACT_ + n];
    }
}

// ---------------------------------------------------------------------------
// Elementwise / staging kernels
// ---------------------------------------------------------------------------
__global__ void cvt_kernel(const float* __restrict__ src, bf16_t* __restrict__ dst,
                           long rows, int src_ld, int dst_ld, int width, int relu)
{
    long total = rows * dst_ld;
    for (long i = (long)blockIdx.x * blockDim.x + threadIdx.x; i < total;
         i += (long)gridDim.x * blockDim.x) {
        long r = i / dst_ld; int c = (int)(i % dst_ld);
        float v = 0.f;
        if (c < width) { v = src[r * src_ld + c]; if (relu) v = fmaxf(v, 0.f); }
        dst[i] = (bf16_t)v;
    }
}

__global__ void whead_kernel(const float* __restrict__ Wa, const float* __restrict__ Wc,
                             bf16_t* __restrict__ dst)
{
    int i = blockIdx.x * blockDim.x + threadIdx.x;
    if (i >= 32 * HID_) return;
    int r = i / HID_, c = i % HID_;
    float v = 0.f;
    if (r < 16) v = Wa[r * HID_ + c];
    else if (r == 16) v = Wc[c];
    dst[i] = (bf16_t)v;
}

__global__ void init_kernel(bf16_t* __restrict__ mi, int* __restrict__ flag)
{
    int i = blockIdx.x * blockDim.x + threadIdx.x;
    if (i == 0) *flag = 1;
    if (i < N_ * 16) {               // zero mi K-pad cols [528,544)
        int n = i / 16, j = i % 16;
        mi[(long)n * KMI_ + (EMB_ + NACT_) + j] = (bf16_t)0.0f;
    }
}

__global__ void allzero_kernel(const float* __restrict__ rnn, int* __restrict__ flag)
{
    long total = (long)N_ * HX_;
    for (long i = (long)blockIdx.x * blockDim.x + threadIdx.x; i < total;
         i += (long)gridDim.x * blockDim.x)
        if (rnn[i] != 0.f) { atomicAnd(flag, 0); return; }
}

__global__ void select_state_kernel(const float* __restrict__ rnn,
                                    bf16_t* __restrict__ mi,
                                    const int* __restrict__ flag)
{
    int i = blockIdx.x * blockDim.x + threadIdx.x;
    if (i >= N_ * EMB_) return;
    if (*flag == 0) {                // not "new": use stored hx_state
        int n = i / EMB_, k = i % EMB_;
        mi[(long)n * KMI_ + k] = (bf16_t)rnn[(long)n * HX_ + 18 + k];
    }
}

__global__ void set_mie_kernel(const bf16_t* __restrict__ miE,
                               bf16_t* __restrict__ mi, int t)
{
    int i = blockIdx.x * blockDim.x + threadIdx.x;
    if (i >= N_ * NACT_) return;
    int n = i / NACT_, j = i % NACT_;
    mi[(long)n * KMI_ + EMB_ + j] = miE[((long)t * N_ + n) * NACT_ + j];
}

__global__ void h_init_kernel(const float* __restrict__ rnn,
                              float* __restrict__ hf, bf16_t* __restrict__ hb)
{
    long i = (long)blockIdx.x * blockDim.x + threadIdx.x;
    if (i >= 2L * N_ * HID_) return;
    int l = (int)(i / ((long)N_ * HID_));
    int n = (int)((i / HID_) % N_);
    int j = (int)(i % HID_);
    float v = rnn[(long)n * HX_ + 530 + l * HID_ + j];
    hf[i] = v; hb[i] = (bf16_t)v;
}

__global__ void hs_copy_kernel(const float* __restrict__ rnn, float* __restrict__ out)
{
    long total = (long)T_ * N_ * (2 * HID_);
    for (long i = (long)blockIdx.x * blockDim.x + threadIdx.x; i < total;
         i += (long)gridDim.x * blockDim.x) {
        int j = (int)(i % (2 * HID_));
        int n = (int)((i / (2 * HID_)) % N_);
        int t = (int)(i / ((long)(2 * HID_) * N_));
        out[((long)t * N_ + n) * HX_ + 530 + j] = rnn[(long)n * HX_ + 530 + j];
    }
}

__global__ void gru_combine_kernel(const float* __restrict__ gi,
                                   const float* __restrict__ gh,
                                   float* __restrict__ hf, bf16_t* __restrict__ hb)
{
    long i = (long)blockIdx.x * blockDim.x + threadIdx.x;
    if (i >= (long)N_ * HID_) return;
    int row = (int)(i / HID_), col = (int)(i % HID_);
    const float* gir = gi + (long)row * G3_;
    const float* ghr = gh + (long)row * G3_;
    float r  = 1.f / (1.f + __expf(-(gir[col]            + ghr[col])));
    float z  = 1.f / (1.f + __expf(-(gir[HID_ + col]     + ghr[HID_ + col])));
    float nn = tanhf(gir[2 * HID_ + col] + r * ghr[2 * HID_ + col]);
    float h  = hf[i];
    float hn = (1.f - z) * nn + z * h;
    hf[i] = hn; hb[i] = (bf16_t)hn;
}

__global__ void copy_last_kernel(float* __restrict__ out)
{
    long i = (long)blockIdx.x * blockDim.x + threadIdx.x;
    if (i >= (long)N_ * HX_) return;
    out[(long)T_ * N_ * HX_ + i] = out[(long)(T_ - 1) * N_ * HX_ + i];
}

// ---------------------------------------------------------------------------
// Launcher
// ---------------------------------------------------------------------------
extern "C" void kernel_launch(void* const* d_in, const int* in_sizes, int n_in,
                              void* d_out, int out_size, void* d_ws, size_t ws_size,
                              hipStream_t stream)
{
    (void)in_sizes; (void)n_in; (void)out_size; (void)ws_size;
    const float* inputs = (const float*)d_in[0];
    const float* rnn    = (const float*)d_in[1];
    const float* W1  = (const float*)d_in[2];  const float* b1   = (const float*)d_in[3];
    const float* W2  = (const float*)d_in[4];  const float* b2   = (const float*)d_in[5];
    const float* W3  = (const float*)d_in[6];  const float* b3   = (const float*)d_in[7];
    const float* Wa  = (const float*)d_in[8];  const float* ba   = (const float*)d_in[9];
    const float* Wc  = (const float*)d_in[10]; const float* bc   = (const float*)d_in[11];
    const float* Eact = (const float*)d_in[12];
    const float* Wih0 = (const float*)d_in[13]; const float* Whh0 = (const float*)d_in[14];
    const float* bih0 = (const float*)d_in[15]; const float* bhh0 = (const float*)d_in[16];
    const float* Wih1 = (const float*)d_in[17]; const float* Whh1 = (const float*)d_in[18];
    const float* bih1 = (const float*)d_in[19]; const float* bhh1 = (const float*)d_in[20];
    float* out = (float*)d_out;

    // --- workspace carve-up (bf16 staging; ~112 MB, L2-resident weights) ---
    char* w = (char*)d_ws;
    auto carve = [&](size_t bytes) -> char* {
        char* p = w; w += (bytes + 255) & ~(size_t)255; return p;
    };
    bf16_t* obsb  = (bf16_t*)carve((size_t)T_ * N_ * OBS_ * 2);   // relu(obs)
    bf16_t* H1b   = (bf16_t*)carve((size_t)T_ * N_ * HID_ * 2);
    bf16_t* Xb    = (bf16_t*)carve((size_t)T_ * N_ * HID_ * 2);   // embed1 out
    bf16_t* W1b   = (bf16_t*)carve((size_t)HID_ * OBS_ * 2);
    bf16_t* W2b   = (bf16_t*)carve((size_t)HID_ * HID_ * 2);
    bf16_t* W3b   = (bf16_t*)carve((size_t)EMB_ * HID_ * 2);
    bf16_t* Whb   = (bf16_t*)carve((size_t)32 * HID_ * 2);
    bf16_t* EactB = (bf16_t*)carve((size_t)NACT_ * NACT_ * 2);
    bf16_t* Wih0b = (bf16_t*)carve((size_t)G3_ * KMI_ * 2);
    bf16_t* Whh0b = (bf16_t*)carve((size_t)G3_ * HID_ * 2);
    bf16_t* Wih1b = (bf16_t*)carve((size_t)G3_ * HID_ * 2);
    bf16_t* Whh1b = (bf16_t*)carve((size_t)G3_ * HID_ * 2);
    bf16_t* miE   = (bf16_t*)carve((size_t)T_ * N_ * NACT_ * 2);
    bf16_t* mi    = (bf16_t*)carve((size_t)N_ * KMI_ * 2);
    float*  gi    = (float*)carve((size_t)N_ * G3_ * 4);
    float*  gh    = (float*)carve((size_t)N_ * G3_ * 4);
    float*  hf    = (float*)carve((size_t)2 * N_ * HID_ * 4);
    bf16_t* hb    = (bf16_t*)carve((size_t)2 * N_ * HID_ * 2);
    int*    flag  = (int*)carve(256);

    dim3 b256(256), b128(128);

    // --- init + "new episode" detection -----------------------------------
    init_kernel<<<16, b256, 0, stream>>>(mi, flag);
    allzero_kernel<<<512, b256, 0, stream>>>(rnn, flag);

    // --- f32 -> bf16 conversions (one pass; weights then live in L2) ------
    cvt_kernel<<<2048, b256, 0, stream>>>(inputs, obsb, (long)T_ * N_, OBS_ + 1, OBS_, OBS_, 1);
    cvt_kernel<<<512,  b256, 0, stream>>>(W1,  W1b,  HID_, OBS_, OBS_, OBS_, 0);
    cvt_kernel<<<1024, b256, 0, stream>>>(W2,  W2b,  HID_, HID_, HID_, HID_, 0);
    cvt_kernel<<<512,  b256, 0, stream>>>(W3,  W3b,  EMB_, HID_, HID_, HID_, 0);
    cvt_kernel<<<1,    b256, 0, stream>>>(Eact, EactB, NACT_, NACT_, NACT_, NACT_, 0);
    cvt_kernel<<<1024, b256, 0, stream>>>(Wih0, Wih0b, G3_, EMB_ + NACT_, KMI_, EMB_ + NACT_, 0);
    cvt_kernel<<<2048, b256, 0, stream>>>(Whh0, Whh0b, G3_, HID_, HID_, HID_, 0);
    cvt_kernel<<<2048, b256, 0, stream>>>(Wih1, Wih1b, G3_, HID_, HID_, HID_, 0);
    cvt_kernel<<<2048, b256, 0, stream>>>(Whh1, Whh1b, G3_, HID_, HID_, HID_, 0);
    whead_kernel<<<(32 * HID_) / 256, b256, 0, stream>>>(Wa, Wc, Whb);

    h_init_kernel<<<(2 * N_ * HID_) / 256, b256, 0, stream>>>(rnn, hf, hb);
    hs_copy_kernel<<<4096, b256, 0, stream>>>(rnn, out);

    // --- time-parallel trunk: embed1 over all T*N rows --------------------
    gemm_wmma_kernel<0, 1><<<dim3(HID_ / 64, (T_ * N_) / 64), b128, 0, stream>>>(
        obsb, OBS_, W1b, OBS_, b1, nullptr, 0, H1b, HID_, OBS_);
    gemm_wmma_kernel<0, 0><<<dim3(HID_ / 64, (T_ * N_) / 64), b128, 0, stream>>>(
        H1b, HID_, W2b, HID_, b2, nullptr, 0, Xb, HID_, HID_);
    head_kernel<<<(T_ * N_) / 64, b128, 0, stream>>>(Xb, Whb, ba, bc, inputs, EactB, out, miE);

    // state0 candidate = embed2(X[0]) -> mi[:, :512]; keep hx_state if !new
    gemm_wmma_kernel<1, 0><<<dim3(EMB_ / 64, N_ / 64), b128, 0, stream>>>(
        Xb, HID_, W3b, HID_, b3, nullptr, 0, mi, KMI_, HID_);
    select_state_kernel<<<(N_ * EMB_) / 256, b256, 0, stream>>>(rnn, mi, flag);

    // --- sequential GRU chain (LDS-staged async WMMA GEMMs) ----------------
    for (int t = 0; t < T_; ++t) {
        set_mie_kernel<<<(N_ * NACT_ + 255) / 256, b256, 0, stream>>>(miE, mi, t);
        // layer 0 gates
        gemm_lds_kernel<<<dim3(G3_ / 64, N_ / 64), b128, 0, stream>>>(
            mi, KMI_, Wih0b, KMI_, bih0, gi, G3_, KMI_);
        gemm_lds_kernel<<<dim3(G3_ / 64, N_ / 64), b128, 0, stream>>>(
            hb, HID_, Whh0b, HID_, bhh0, gh, G3_, HID_);
        gru_combine_kernel<<<(N_ * HID_) / 256, b256, 0, stream>>>(gi, gh, hf, hb);
        // layer 1 gates
        gemm_lds_kernel<<<dim3(G3_ / 64, N_ / 64), b128, 0, stream>>>(
            hb, HID_, Wih1b, HID_, bih1, gi, G3_, HID_);
        gemm_lds_kernel<<<dim3(G3_ / 64, N_ / 64), b128, 0, stream>>>(
            hb + (long)N_ * HID_, HID_, Whh1b, HID_, bhh1, gh, G3_, HID_);
        gru_combine_kernel<<<(N_ * HID_) / 256, b256, 0, stream>>>(
            gi, gh, hf + (long)N_ * HID_, hb + (long)N_ * HID_);
        // new_state = embed2(h1n): f32 -> out Ss slice, bf16 -> next-step mi
        gemm_wmma_kernel<1, 0><<<dim3(EMB_ / 64, N_ / 64), b128, 0, stream>>>(
            hb + (long)N_ * HID_, HID_, W3b, HID_, b3,
            out + (long)t * N_ * HX_ + 18, HX_, mi, KMI_, HID_);
    }

    // hx[-1:] tail
    copy_last_kernel<<<(N_ * HX_) / 256, b256, 0, stream>>>(out);
}
